// GraphormerBlock_22016002359711
// MI455X (gfx1250) — matrix operand order repaired and checked by reading, not compiled, not run on hardware
//
#include <hip/hip_runtime.h>
#include <hip/hip_bf16.h>
#include <math.h>

// ---------------------------------------------------------------------------
// GraphormerBlock for MI455X (gfx1250, wave32).
// All GEMMs via v_wmma_f32_16x16x32_bf16 (bf16 in, f32 accumulate).
// Weights are repacked transposed+bf16 once per launch so:
//   A fragment (16x32): lane L holds row (L&15); half j -> K = k0+16*(L>>4)+j
//                       (two 16B contiguous loads)
//   B fragment (32x16): lane L holds col (L&15); half j -> K = k0+16*(L>>4)+j
//                       (one 32B contiguous load from the transposed weight)
//   C/D (16x16 f32):    lane L col (L&15); vgpr i -> row i+8*(L>>4)
// Workspace requirement: ~1.02 GB.
// ---------------------------------------------------------------------------

#define NN 50000
#define EE 800000
#define DIM 128
#define EDIM 64
#define HH 8

typedef __attribute__((ext_vector_type(16))) __bf16 v16bf;
typedef __attribute__((ext_vector_type(8)))  __bf16 v8bf;
typedef __attribute__((ext_vector_type(4)))  __bf16 v4bf;
typedef __attribute__((ext_vector_type(8)))  float  v8f;

__device__ __forceinline__ v16bf cat8(v8bf lo, v8bf hi) {
  return __builtin_shufflevector(lo, hi, 0,1,2,3,4,5,6,7,8,9,10,11,12,13,14,15);
}

__device__ __forceinline__ float wsum(float v) {
#pragma unroll
  for (int o = 16; o; o >>= 1) v += __shfl_xor(v, o, 32);
  return v;
}

__device__ __forceinline__ void atomicMaxF(float* a, float v) {
  if (v >= 0.f) atomicMax((int*)a, __float_as_int(v));
  else          atomicMin((unsigned int*)a, __float_as_uint(v));
}

// ---------------------------------------------------------------- init / misc
__global__ void k_init(float* agg, float* amax, float* denom, int* deg) {
  long t = (long)blockIdx.x * blockDim.x + threadIdx.x;
  if (t < (long)NN * DIM) agg[t] = 0.f;
  if (t < (long)NN * HH) { amax[t] = -__builtin_inff(); denom[t] = 0.f; }
  if (t < NN) deg[t] = 0;
}

__global__ void k_deg(const int* __restrict__ src, int* deg) {
  long t = (long)blockIdx.x * blockDim.x + threadIdx.x;
  if (t < EE) atomicAdd(&deg[src[t]], 1);
}

__global__ void k_cvt_ea(const float* __restrict__ ea, __bf16* __restrict__ out) {
  long t = (long)blockIdx.x * blockDim.x + threadIdx.x;
  if (t < (long)EE * EDIM) out[t] = (__bf16)ea[t];
}

// Transpose + convert all weight matrices to bf16 (B operand layout).
__global__ void k_pack(const float* Wq, const float* Wk, const float* Wv, const float* Ws,
                       const float* bq, const float* bk, const float* bv, const float* bs,
                       const float* We, const float* W1, const float* W2, const float* euW,
                       __bf16* wcat, float* bcat, __bf16* wet, __bf16* w1t,
                       __bf16* w2t, __bf16* euwt) {
  int o = blockIdx.x * blockDim.x + threadIdx.x;
  if (o < 512 * 128) {           // wcat[n][k] : n = q|k|v|skip columns
    int n = o / 128, k = o % 128;
    const float* S = (n < 128) ? Wq : (n < 256) ? Wk : (n < 384) ? Wv : Ws;
    wcat[o] = (__bf16)S[k * 128 + (n & 127)];
    return;
  }
  o -= 512 * 128;
  if (o < 512) {
    const float* S = (o < 128) ? bq : (o < 256) ? bk : (o < 384) ? bv : bs;
    bcat[o] = S[o & 127];
    return;
  }
  o -= 512;
  if (o < 128 * 64) { int n = o / 64, k = o % 64; wet[o] = (__bf16)We[k * 128 + n]; return; }
  o -= 128 * 64;
  if (o < 512 * 128) { int n = o / 128, k = o % 128; w1t[o] = (__bf16)W1[k * 512 + n]; return; }
  o -= 512 * 128;
  if (o < 128 * 512) { int n = o / 512, k = o % 512; w2t[o] = (__bf16)W2[k * 128 + n]; return; }
  o -= 128 * 512;
  if (o < 128 * 320) { int n = o / 320, k = o % 320; euwt[o] = (__bf16)euW[k * 128 + n]; return; }
}

// ----------------------------------------------------------- node pre (wave/row)
__global__ void k_node_pre(const float* __restrict__ x, const int* __restrict__ deg,
                           const float* __restrict__ demb, const float* __restrict__ g,
                           const float* __restrict__ b, float* __restrict__ x_in,
                           __bf16* __restrict__ hbf) {
  int w = (blockIdx.x * blockDim.x + threadIdx.x) >> 5;
  int lane = threadIdx.x & 31;
  if (w >= NN) return;
  int dg = deg[w]; dg = dg > 255 ? 255 : dg;
  float4 xv = ((const float4*)(x + (long)w * DIM))[lane];
  float4 ev = ((const float4*)(demb + (long)dg * DIM))[lane];
  float e0 = xv.x + ev.x, e1 = xv.y + ev.y, e2 = xv.z + ev.z, e3 = xv.w + ev.w;
  float4 xi = {e0, e1, e2, e3};
  ((float4*)(x_in + (long)w * DIM))[lane] = xi;
  float m = wsum(e0 + e1 + e2 + e3) * (1.f / DIM);
  float d0 = e0 - m, d1 = e1 - m, d2 = e2 - m, d3 = e3 - m;
  float vv = wsum(d0 * d0 + d1 * d1 + d2 * d2 + d3 * d3) * (1.f / DIM);
  float r = rsqrtf(vv + 1e-5f);
  int j = lane * 4;
  v4bf h;
  h[0] = (__bf16)(d0 * r * g[j + 0] + b[j + 0]);
  h[1] = (__bf16)(d1 * r * g[j + 1] + b[j + 1]);
  h[2] = (__bf16)(d2 * r * g[j + 2] + b[j + 2]);
  h[3] = (__bf16)(d3 * r * g[j + 3] + b[j + 3]);
  ((v4bf*)(hbf + (long)w * DIM))[lane] = h;
}

// --------------------------------------------------------------- generic GEMM
// C[M x Nc] = A[M x K](bf16) @ Wt[Nc x K]^T(bf16) (+bias)(+gelu)(+resid)
// One wave computes a 16x64 output strip (4 accumulators).
__global__ void k_gemm(const __bf16* __restrict__ A, const __bf16* __restrict__ Wt,
                       const float* __restrict__ bias, const float* __restrict__ resid,
                       float* __restrict__ outF, __bf16* __restrict__ outBF,
                       int M, int Nc, int K, int gelu) {
  int lane = threadIdx.x & 31;
  int wid = blockIdx.x * (blockDim.x >> 5) + (threadIdx.x >> 5);
  int tilesN = Nc >> 6;
  int totalW = (M >> 4) * tilesN;
  if (wid >= totalW) return;                       // wave-uniform guard
  int rowTile = wid / tilesN, colChunk = wid - rowTile * tilesN;
  int row0 = rowTile << 4, col0 = colChunk << 6;
  int hs = lane >> 4, rA = lane & 15, nB = lane & 15;
  const __bf16* Ab = A + (long)(row0 + rA) * K + 16 * hs;
  const __bf16* Bb = Wt + (long)(col0 + nB) * K + 16 * hs;
  long bstr = (long)16 * K;
  v8f z = {0.f, 0.f, 0.f, 0.f, 0.f, 0.f, 0.f, 0.f};
  v8f acc0 = z, acc1 = z, acc2 = z, acc3 = z;
  for (int k0 = 0; k0 < K; k0 += 32) {
    __builtin_prefetch(Ab + k0 + 64, 0, 1);        // stream A ahead
    v8bf alo = *(const v8bf*)(Ab + k0);
    v8bf ahi = *(const v8bf*)(Ab + k0 + 8);
    v16bf a = cat8(alo, ahi);
    v16bf b0 = *(const v16bf*)(Bb + k0);
    v16bf b1 = *(const v16bf*)(Bb + bstr + k0);
    v16bf b2 = *(const v16bf*)(Bb + 2 * bstr + k0);
    v16bf b3 = *(const v16bf*)(Bb + 3 * bstr + k0);
    acc0 = __builtin_amdgcn_wmma_f32_16x16x32_bf16(false, a, false, b0, (short)0, acc0, false, false);
    acc1 = __builtin_amdgcn_wmma_f32_16x16x32_bf16(false, a, false, b1, (short)0, acc1, false, false);
    acc2 = __builtin_amdgcn_wmma_f32_16x16x32_bf16(false, a, false, b2, (short)0, acc2, false, false);
    acc3 = __builtin_amdgcn_wmma_f32_16x16x32_bf16(false, a, false, b3, (short)0, acc3, false, false);
  }
  v8f accs[4] = {acc0, acc1, acc2, acc3};
  int m0 = row0 + 8 * hs;
#pragma unroll
  for (int t = 0; t < 4; t++) {
    int n = col0 + t * 16 + nB;
    float bv = bias ? bias[n] : 0.f;
#pragma unroll
    for (int i = 0; i < 8; i++) {
      long idx = (long)(m0 + i) * Nc + n;
      float v = accs[t][i] + bv;
      if (gelu) v = 0.5f * v * (1.f + erff(v * 0.70710678118654752f));
      if (resid) v += resid[idx];
      if (outF) outF[idx] = v;
      else outBF[idx] = (__bf16)v;
    }
  }
}

// ------------------------------------------------------- attention edge passes
__global__ void k_edge_alpha(const float* __restrict__ qkvs, const __bf16* __restrict__ ebf,
                             const int* __restrict__ src, const int* __restrict__ dst,
                             float* __restrict__ alpha, float* __restrict__ amax) {
  long t = (long)blockIdx.x * blockDim.x + threadIdx.x;
  if (t >= (long)EE * HH) return;
  int e = (int)(t >> 3), hd = (int)(t & 7);
  int s = src[e], d = dst[e];
  const float*  q = qkvs + (long)d * 512 + hd * 16;
  const float*  k = qkvs + (long)s * 512 + 128 + hd * 16;
  const __bf16* ee = ebf + (long)e * DIM + hd * 16;
  float acc = 0.f;
#pragma unroll
  for (int c = 0; c < 16; c++) acc += q[c] * (k[c] + (float)ee[c]);
  acc *= 0.25f;                                    // 1/sqrt(C), C=16
  alpha[t] = acc;
  atomicMaxF(amax + (long)d * HH + hd, acc);
}

__global__ void k_edge_scatter(const float* __restrict__ qkvs, const __bf16* __restrict__ ebf,
                               const float* __restrict__ alpha, const float* __restrict__ amax,
                               float* __restrict__ denom, float* __restrict__ agg,
                               const int* __restrict__ src, const int* __restrict__ dst) {
  long t = (long)blockIdx.x * blockDim.x + threadIdx.x;
  if (t >= (long)EE * HH) return;
  int e = (int)(t >> 3), hd = (int)(t & 7);
  int s = src[e], d = dst[e];
  float pe = expf(alpha[t] - amax[(long)d * HH + hd]);
  atomicAdd(denom + (long)d * HH + hd, pe);
  const float*  v = qkvs + (long)s * 512 + 256 + hd * 16;
  const __bf16* ee = ebf + (long)e * DIM + hd * 16;
  float* ag = agg + (long)d * DIM + hd * 16;
#pragma unroll
  for (int c = 0; c < 16; c++) atomicAdd(ag + c, pe * (v[c] + (float)ee[c]));
}

// ----------------------------------------------------------- node post (wave/row)
__global__ void k_node_post(const float* __restrict__ x_in, const float* __restrict__ agg,
                            const float* __restrict__ denom, const float* __restrict__ qkvs,
                            const float* __restrict__ bskip, const float* __restrict__ g,
                            const float* __restrict__ b, float* __restrict__ x_att,
                            __bf16* __restrict__ h2bf) {
  int w = (blockIdx.x * blockDim.x + threadIdx.x) >> 5;
  int lane = threadIdx.x & 31;
  if (w >= NN) return;
  float dn = denom[(long)w * HH + (lane >> 2)];
  float inv = dn > 0.f ? 1.f / dn : 0.f;
  float4 ag = ((const float4*)(agg + (long)w * DIM))[lane];
  float4 xi = ((const float4*)(x_in + (long)w * DIM))[lane];
  float4 sk = ((const float4*)(qkvs + (long)w * 512 + 384))[lane];
  float4 bs = ((const float4*)bskip)[lane];
  float e0 = xi.x + ag.x * inv + sk.x + bs.x;
  float e1 = xi.y + ag.y * inv + sk.y + bs.y;
  float e2 = xi.z + ag.z * inv + sk.z + bs.z;
  float e3 = xi.w + ag.w * inv + sk.w + bs.w;
  float4 xa = {e0, e1, e2, e3};
  ((float4*)(x_att + (long)w * DIM))[lane] = xa;
  float m = wsum(e0 + e1 + e2 + e3) * (1.f / DIM);
  float d0 = e0 - m, d1 = e1 - m, d2 = e2 - m, d3 = e3 - m;
  float vv = wsum(d0 * d0 + d1 * d1 + d2 * d2 + d3 * d3) * (1.f / DIM);
  float r = rsqrtf(vv + 1e-5f);
  int j = lane * 4;
  v4bf h;
  h[0] = (__bf16)(d0 * r * g[j + 0] + b[j + 0]);
  h[1] = (__bf16)(d1 * r * g[j + 1] + b[j + 1]);
  h[2] = (__bf16)(d2 * r * g[j + 2] + b[j + 2]);
  h[3] = (__bf16)(d3 * r * g[j + 3] + b[j + 3]);
  ((v4bf*)(h2bf + (long)w * DIM))[lane] = h;
}

// LN(d_out x_new) -> xn_bf, for edge update gathers
__global__ void k_ln128(const float* __restrict__ in, const float* __restrict__ g,
                        const float* __restrict__ b, __bf16* __restrict__ out) {
  int w = (blockIdx.x * blockDim.x + threadIdx.x) >> 5;
  int lane = threadIdx.x & 31;
  if (w >= NN) return;
  float4 xv = ((const float4*)(in + (long)w * DIM))[lane];
  float m = wsum(xv.x + xv.y + xv.z + xv.w) * (1.f / DIM);
  float d0 = xv.x - m, d1 = xv.y - m, d2 = xv.z - m, d3 = xv.w - m;
  float vv = wsum(d0 * d0 + d1 * d1 + d2 * d2 + d3 * d3) * (1.f / DIM);
  float r = rsqrtf(vv + 1e-5f);
  int j = lane * 4;
  v4bf h;
  h[0] = (__bf16)(d0 * r * g[j + 0] + b[j + 0]);
  h[1] = (__bf16)(d1 * r * g[j + 1] + b[j + 1]);
  h[2] = (__bf16)(d2 * r * g[j + 2] + b[j + 2]);
  h[3] = (__bf16)(d3 * r * g[j + 3] + b[j + 3]);
  ((v4bf*)(out + (long)w * DIM))[lane] = h;
}

// -------------------------------------------- edge-update GEMM with gathered A
// raw[E x 128] = relu( [xn[src] | xn[dst] | ea] (E x 320) @ euWt^T + eu_b )
__device__ __forceinline__ v8bf gather8(const __bf16* __restrict__ xn,
                                        const __bf16* __restrict__ ea,
                                        int s, int d, long e, int kb) {
  const __bf16* p;
  if (kb < 128)      p = xn + (long)s * DIM + kb;
  else if (kb < 256) p = xn + (long)d * DIM + (kb - 128);
  else               p = ea + e * EDIM + (kb - 256);
  return *(const v8bf*)p;
}

__global__ void k_gemm_eu(const __bf16* __restrict__ xn, const __bf16* __restrict__ ea,
                          const int* __restrict__ src, const int* __restrict__ dst,
                          const __bf16* __restrict__ euwt, const float* __restrict__ eub,
                          float* __restrict__ raw) {
  const int Nc = 128, K = 320;
  int lane = threadIdx.x & 31;
  int wid = blockIdx.x * (blockDim.x >> 5) + (threadIdx.x >> 5);
  const int tilesN = Nc >> 6;
  if (wid >= (EE >> 4) * tilesN) return;
  int rowTile = wid / tilesN, colChunk = wid - rowTile * tilesN;
  int row0 = rowTile << 4, col0 = colChunk << 6;
  int hs = lane >> 4, rA = lane & 15, nB = lane & 15;
  long eRow = row0 + rA;
  int s = src[eRow], d = dst[eRow];
  const __bf16* Bb = euwt + (long)(col0 + nB) * K + 16 * hs;
  long bstr = (long)16 * K;
  v8f z = {0.f, 0.f, 0.f, 0.f, 0.f, 0.f, 0.f, 0.f};
  v8f acc0 = z, acc1 = z, acc2 = z, acc3 = z;
#pragma unroll
  for (int k0 = 0; k0 < K; k0 += 32) {
    int kb = k0 + 16 * hs;
    v16bf a = cat8(gather8(xn, ea, s, d, eRow, kb),
                   gather8(xn, ea, s, d, eRow, kb + 8));
    v16bf b0 = *(const v16bf*)(Bb + k0);
    v16bf b1 = *(const v16bf*)(Bb + bstr + k0);
    v16bf b2 = *(const v16bf*)(Bb + 2 * bstr + k0);
    v16bf b3 = *(const v16bf*)(Bb + 3 * bstr + k0);
    acc0 = __builtin_amdgcn_wmma_f32_16x16x32_bf16(false, a, false, b0, (short)0, acc0, false, false);
    acc1 = __builtin_amdgcn_wmma_f32_16x16x32_bf16(false, a, false, b1, (short)0, acc1, false, false);
    acc2 = __builtin_amdgcn_wmma_f32_16x16x32_bf16(false, a, false, b2, (short)0, acc2, false, false);
    acc3 = __builtin_amdgcn_wmma_f32_16x16x32_bf16(false, a, false, b3, (short)0, acc3, false, false);
  }
  v8f accs[4] = {acc0, acc1, acc2, acc3};
  int m0 = row0 + 8 * hs;
#pragma unroll
  for (int t = 0; t < 4; t++) {
    int n = col0 + t * 16 + nB;
    float bv = eub[n];
#pragma unroll
    for (int i = 0; i < 8; i++) {
      float v = accs[t][i] + bv;
      raw[(long)(m0 + i) * Nc + n] = v > 0.f ? v : 0.f;   // relu fused
    }
  }
}

// edge_new = LN(ea + sigmoid(gate)*delta), wave per edge (64 feats, 2/lane)
__global__ void k_edge_final(const float* __restrict__ raw, const float* __restrict__ ea,
                             const float* __restrict__ g, const float* __restrict__ b,
                             float* __restrict__ out) {
  int w = (blockIdx.x * blockDim.x + threadIdx.x) >> 5;
  int lane = threadIdx.x & 31;
  if (w >= EE) return;
  const float* r = raw + (long)w * 128;
  float2 dv = ((const float2*)r)[lane];
  float2 gv = ((const float2*)(r + 64))[lane];
  float2 av = ((const float2*)(ea + (long)w * EDIM))[lane];
  float e0 = av.x + dv.x / (1.f + expf(-gv.x));
  float e1 = av.y + dv.y / (1.f + expf(-gv.y));
  float m = wsum(e0 + e1) * (1.f / 64.f);
  float d0 = e0 - m, d1 = e1 - m;
  float vv = wsum(d0 * d0 + d1 * d1) * (1.f / 64.f);
  float rr = rsqrtf(vv + 1e-5f);
  int j = lane * 2;
  float2 o = {d0 * rr * g[j] + b[j], d1 * rr * g[j + 1] + b[j + 1]};
  ((float2*)(out + (long)w * EDIM))[lane] = o;
}

// ---------------------------------------------------------------------------
extern "C" void kernel_launch(void* const* d_in, const int* in_sizes, int n_in,
                              void* d_out, int out_size, void* d_ws, size_t ws_size,
                              hipStream_t stream) {
  (void)in_sizes; (void)n_in; (void)out_size; (void)ws_size;
  const float* x    = (const float*)d_in[0];
  const int*   eidx = (const int*)d_in[1];
  const float* ea   = (const float*)d_in[2];
  const float* demb = (const float*)d_in[3];
  const float* ng   = (const float*)d_in[4];
  const float* nb   = (const float*)d_in[5];
  const float* Wq   = (const float*)d_in[6];  const float* bq = (const float*)d_in[7];
  const float* Wk   = (const float*)d_in[8];  const float* bk = (const float*)d_in[9];
  const float* Wv   = (const float*)d_in[10]; const float* bv = (const float*)d_in[11];
  const float* We   = (const float*)d_in[12];
  const float* Ws   = (const float*)d_in[13]; const float* bs = (const float*)d_in[14];
  const float* W1   = (const float*)d_in[15]; const float* b1 = (const float*)d_in[16];
  const float* W2   = (const float*)d_in[17]; const float* b2 = (const float*)d_in[18];
  const float* eung = (const float*)d_in[19]; const float* eunb = (const float*)d_in[20];
  const float* euW  = (const float*)d_in[21]; const float* eub  = (const float*)d_in[22];
  const float* eueg = (const float*)d_in[23]; const float* eueb = (const float*)d_in[24];
  const int* src = eidx;
  const int* dst = eidx + EE;
  float* xout = (float*)d_out;                 // [N,128]
  float* eout = xout + (long)NN * DIM;         // [E,64]

  char* base = (char*)d_ws;
  size_t off = 0;
  auto alloc = [&](size_t bytes) -> void* {
    void* p = base + off;
    off = (off + bytes + 255) & ~(size_t)255;
    return p;
  };
  int*    deg   = (int*)   alloc((size_t)NN * 4);
  float*  x_in  = (float*) alloc((size_t)NN * DIM * 4);
  __bf16* hbf   = (__bf16*)alloc((size_t)NN * DIM * 2);
  float*  qkvs  = (float*) alloc((size_t)NN * 512 * 4);
  __bf16* eabf  = (__bf16*)alloc((size_t)EE * EDIM * 2);
  __bf16* ebf   = (__bf16*)alloc((size_t)EE * DIM * 2);
  float*  alpha = (float*) alloc((size_t)EE * HH * 4);
  float*  amax  = (float*) alloc((size_t)NN * HH * 4);
  float*  denom = (float*) alloc((size_t)NN * HH * 4);
  float*  agg   = (float*) alloc((size_t)NN * DIM * 4);
  float*  x_att = (float*) alloc((size_t)NN * DIM * 4);
  __bf16* h2bf  = (__bf16*)alloc((size_t)NN * DIM * 2);
  __bf16* midbf = (__bf16*)alloc((size_t)NN * 512 * 2);
  __bf16* xnbf  = (__bf16*)alloc((size_t)NN * DIM * 2);
  float*  raw   = (float*) alloc((size_t)EE * DIM * 4);
  __bf16* wcat  = (__bf16*)alloc(512 * 128 * 2);
  float*  bcat  = (float*) alloc(512 * 4);
  __bf16* wet   = (__bf16*)alloc(128 * 64 * 2);
  __bf16* w1t   = (__bf16*)alloc(512 * 128 * 2);
  __bf16* w2t   = (__bf16*)alloc(128 * 512 * 2);
  __bf16* euwt  = (__bf16*)alloc(128 * 320 * 2);

  const int B = 256;
  // init + degree + conversions + weight pack
  k_init<<<(NN * DIM + B - 1) / B, B, 0, stream>>>(agg, amax, denom, deg);
  k_deg<<<(EE + B - 1) / B, B, 0, stream>>>(src, deg);
  k_cvt_ea<<<(int)(((long)EE * EDIM + B - 1) / B), B, 0, stream>>>(ea, eabf);
  k_pack<<<(246272 + B - 1) / B, B, 0, stream>>>(Wq, Wk, Wv, Ws, bq, bk, bv, bs,
                                                 We, W1, W2, euW,
                                                 wcat, bcat, wet, w1t, w2t, euwt);
  // node pre-LN
  k_node_pre<<<(NN * 32 + B - 1) / B, B, 0, stream>>>(x, deg, demb, ng, nb, x_in, hbf);
  // fused q|k|v|skip GEMM: [N,128]@[128,512]
  {
    int waves = (NN / 16) * (512 / 64);
    k_gemm<<<(waves + 7) / 8, B, 0, stream>>>(hbf, wcat, bcat, nullptr, qkvs, nullptr,
                                              NN, 512, 128, 0);
  }
  // edge projection: [E,64]@[64,128] -> bf16
  {
    int waves = (EE / 16) * (128 / 64);
    k_gemm<<<(waves + 7) / 8, B, 0, stream>>>(eabf, wet, nullptr, nullptr, nullptr, ebf,
                                              EE, 128, 64, 0);
  }
  // attention: alpha + segment max, then exp + segment sums
  k_edge_alpha<<<(int)(((long)EE * HH + B - 1) / B), B, 0, stream>>>(qkvs, ebf, src, dst, alpha, amax);
  k_edge_scatter<<<(int)(((long)EE * HH + B - 1) / B), B, 0, stream>>>(qkvs, ebf, alpha, amax,
                                                                       denom, agg, src, dst);
  // x_attn + pre-FFN LN
  k_node_post<<<(NN * 32 + B - 1) / B, B, 0, stream>>>(x_in, agg, denom, qkvs, bs, ng, nb,
                                                       x_att, h2bf);
  // FFN1 (gelu fused): [N,128]@[128,512] -> bf16
  {
    int waves = (NN / 16) * (512 / 64);
    k_gemm<<<(waves + 7) / 8, B, 0, stream>>>(h2bf, w1t, b1, nullptr, nullptr, midbf,
                                              NN, 512, 128, 1);
  }
  // FFN2 (+residual) writes x_new straight into d_out: [N,512]@[512,128]
  {
    int waves = (NN / 16) * (128 / 64);
    k_gemm<<<(waves + 7) / 8, B, 0, stream>>>(midbf, w2t, b2, x_att, xout, nullptr,
                                              NN, 128, 512, 0);
  }
  // xn = LN(x_new) for edge update
  k_ln128<<<(NN * 32 + B - 1) / B, B, 0, stream>>>(xout, eung, eunb, xnbf);
  // edge-update GEMM with on-the-fly gather of [xn[src]|xn[dst]|ea]
  {
    int waves = (EE / 16) * (128 / 64);
    k_gemm_eu<<<(waves + 7) / 8, B, 0, stream>>>(xnbf, eabf, src, dst, euwt, eub, raw);
  }
  // gated residual + final LN over edge features
  k_edge_final<<<(EE * 32 + B - 1) / B, B, 0, stream>>>(raw, ea, eueg, eueb, eout);
}
